// DaRnnModel_24034636989058
// MI455X (gfx1250) — matrix operand order repaired and checked by reading, not compiled, not run on hardware
//
#include <hip/hip_runtime.h>
#include <hip/hip_fp16.h>

#define B_ 1024
#define T_ 256
#define N_ 128
#define M_ 256

typedef __attribute__((ext_vector_type(16))) _Float16 v16h;
typedef __attribute__((ext_vector_type(8)))  float    v8f;
typedef __attribute__((ext_vector_type(4)))  unsigned int u32x4;
typedef __attribute__((ext_vector_type(8)))  int      i32x8;
typedef __attribute__((ext_vector_type(4)))  int      i32x4;

// Native v_tanh_f32 is a single TRANS32 op on CDNA5 (co-executes with WMMA);
// fall back to the exp-based form if the builtin is absent.
__device__ __forceinline__ float fast_tanh(float x) {
#if __has_builtin(__builtin_amdgcn_tanhf)
  return __builtin_amdgcn_tanhf(x);
#else
  float e = __expf(2.0f * x);
  return (e - 1.0f) / (e + 1.0f);
#endif
}
__device__ __forceinline__ float fast_sig(float x) {
#if __has_builtin(__builtin_amdgcn_tanhf)
  return 0.5f * (1.0f + __builtin_amdgcn_tanhf(0.5f * x));
#else
  return 1.0f / (1.0f + __expf(-x));
#endif
}

// ---- CDNA5 WMMA 16x16x32 f16 fragment helpers (layouts per CDNA5 ISA) ----
template <typename F>
__device__ __forceinline__ v16h load_frag_a(F src) {
  const int l = threadIdx.x & 31;
  const int row = l & 15;
  const int koff = (l >> 4) << 3;   // 0 or 8
  v16h a;
#pragma unroll
  for (int i = 0; i < 8; ++i) a[i] = src(row, koff + i);
#pragma unroll
  for (int i = 0; i < 8; ++i) a[8 + i] = src(row, koff + 16 + i);
  return a;
}
// Packed-B: weights pre-swizzled so each lane reads 16 contiguous halves (2x b128)
__device__ __forceinline__ v16h load_frag_b_packed(const _Float16* __restrict__ p) {
  return *(const v16h*)(p + (size_t)(threadIdx.x & 31) * 16);
}
template <typename F>
__device__ __forceinline__ void store_frag_c(const v8f& c, F f) {
  const int l = threadIdx.x & 31;
  const int col = l & 15;
  const int rowbase = (l >> 4) << 3;
#pragma unroll
  for (int i = 0; i < 8; ++i) f(rowbase + i, col, c[i]);
}

#define WMMA_F16(a, b, c) \
  __builtin_amdgcn_wmma_f32_16x16x32_f16(false, (a), false, (b), (short)0, (c), false, false)

// ============================================================================
// Kernel 0: pack a row-major f32 weight W[outdim][K] into f16 WMMA-B-fragment
// order: block (ct, kk) -> 32 lanes x 16 contiguous halves.
// frag elem i of lane l = W[ct*16 + (l&15)][kk*32 + ((l>>4)<<3) + (i < 8 ? i : i + 8)]
// ============================================================================
__global__ __launch_bounds__(256) void pack_b_kernel(
    const float* __restrict__ W, _Float16* __restrict__ P, int outtiles,
    int nkk, int K) {
  const int idx = blockIdx.x * 256 + threadIdx.x;
  if (idx >= outtiles * nkk * 32) return;
  const int lane = idx & 31;
  const int kk = (idx >> 5) % nkk;
  const int ct = idx / (32 * nkk);
  const int col = lane & 15;
  const int kbase = kk * 32 + ((lane >> 4) << 3);
  v16h t;
#pragma unroll
  for (int i = 0; i < 16; ++i)
    t[i] = (_Float16)W[(size_t)(ct * 16 + col) * K + kbase + (i < 8 ? i : i + 8)];
  *(v16h*)(P + (size_t)idx * 16) = t;
}

// ============================================================================
// Kernel 1: ue_x[b,n,t] = sum_k x[b,k,n] * Ue_w[t,k] + Ue_b[t]   (stored f16)
// 8 waves/block share one (b, ntile): stage the 256x16 f32 x-slice into LDS
// via the Tensor Data Mover (2D tile: 16 floats/row, 256 rows, stride 128).
// ============================================================================
__global__ __launch_bounds__(256) void ue_gemm_kernel(
    const float* __restrict__ x, const _Float16* __restrict__ pUe,
    const float* __restrict__ Ue_b, _Float16* __restrict__ ue_x) {
  __shared__ float sh_x[T_][16];  // [t'][n] 16 KB

  const int group = blockIdx.x;
  const int ttbase = (group & 1) * 8;
  const int ntile = (group >> 1) & 7;
  const int b = group >> 4;
  const int w = threadIdx.x >> 5;
  const float* xb = x + (size_t)b * T_ * N_ + ntile * 16;

#if __has_builtin(__builtin_amdgcn_tensor_load_to_lds)
  if (threadIdx.x < 32) {
    const unsigned long long ga = (unsigned long long)(uintptr_t)xb;
    const unsigned la = (unsigned)(uintptr_t)&sh_x[0][0];  // generic LDS addr low 32b = offset
    u32x4 g0;
    g0[0] = 1u;                                              // count=1 valid descriptor
    g0[1] = la;                                              // D#.lds_addr
    g0[2] = (unsigned)ga;                                    // D#.global_addr[31:0]
    g0[3] = (unsigned)((ga >> 32) & 0x01FFFFFFu) | (2u << 30);  // addr[56:32] | type=2
    i32x8 g1;
    g1[0] = (2 << 16);          // workgroup_mask=0 | data_size=2 (4 bytes)
    g1[1] = (16 << 16);         // tensor_dim0 = 16 (elements per row)
    g1[2] = (256 << 16);        // tensor_dim1 = 256 rows
    g1[3] = (16 << 16);         // tile_dim0 = 16
    g1[4] = 256;                // tile_dim1 = 256, tile_dim2 = 0
    g1[5] = N_;                 // tensor_dim0_stride = 128 elements
    g1[6] = 0;
    g1[7] = 0;
    i32x4 z4 = {0, 0, 0, 0};            // groups 2/3 unused (2D tensor)
    i32x8 z8 = {0, 0, 0, 0, 0, 0, 0, 0};
    __builtin_amdgcn_tensor_load_to_lds(g0, g1, z4, z4, z8, 0);
#if __has_builtin(__builtin_amdgcn_s_wait_tensorcnt)
    __builtin_amdgcn_s_wait_tensorcnt((short)0);
#else
    asm volatile("s_wait_tensorcnt 0x0" ::: "memory");
#endif
  }
#else
  {  // cooperative fallback: thread tid stages row t'=tid (4x b128)
    const int tid = threadIdx.x;
#pragma unroll
    for (int i = 0; i < 16; ++i) sh_x[tid][i] = xb[(size_t)tid * N_ + i];
  }
#endif
  __syncthreads();

  const int ttile = ttbase + w;
  v8f c = {};
  for (int kk = 0; kk < T_ / 32; ++kk) {
    const int kb = kk * 32;
    v16h a = load_frag_a([&](int r, int k) -> _Float16 {
      return (_Float16)sh_x[kb + k][r];
    });
    v16h bb = load_frag_b_packed(pUe + (size_t)(ttile * 8 + kk) * 512);
    c = WMMA_F16(a, bb, c);
  }
  store_frag_c(c, [&](int r, int col, float v) {
    const int n = ntile * 16 + r;
    const int t = ttile * 16 + col;
    ue_x[((size_t)b * N_ + n) * T_ + t] = (_Float16)(v + Ue_b[t]);
  });
}

// ============================================================================
// Kernel 2: the sequential recurrence. 64 workgroups x 16 batch rows.
// 1024 threads: waves 0-15 do WMMA + fused pointwise (cell state in regs),
// all 32 waves share the trans-op-heavy attention phase.
// ============================================================================
__global__ __launch_bounds__(1024) void darnn_recurrence_kernel(
    const float* __restrict__ x, const _Float16* __restrict__ pWe,
    const float* __restrict__ We_b, const float* __restrict__ ve_w,
    const float* __restrict__ ve_b, const _Float16* __restrict__ pWih,
    const _Float16* __restrict__ pWhh, const float* __restrict__ b_ih,
    const float* __restrict__ b_hh, const _Float16* __restrict__ ue_x,
    _Float16* __restrict__ h_cat) {
  __shared__ _Float16 sh_h[16][M_];     // 8 KB
  __shared__ _Float16 sh_c[16][M_];     // 8 KB
  __shared__ float    sh_we[16][T_];    // 16 KB
  __shared__ float    sh_sc[16][N_];    // 8 KB (e scores -> alpha)
  __shared__ _Float16 sh_xt[16][N_];    // 4 KB (x_tilde)
  __shared__ float    sh_ve[T_];        // 1 KB
  __shared__ float    sh_bias[4 * M_];  // 4 KB

  const int tid = threadIdx.x;
  const int w = tid >> 5;               // wave 0..31
  const int b0 = blockIdx.x * 16;

  for (int i = tid; i < 16 * M_; i += 1024) {
    sh_h[i >> 8][i & 255] = (_Float16)0.f;
    sh_c[i >> 8][i & 255] = (_Float16)0.f;
  }
  if (tid < T_) sh_ve[tid] = ve_w[tid];
  for (int i = tid; i < 4 * M_; i += 1024) sh_bias[i] = b_ih[i] + b_hh[i];
  __syncthreads();

  const float veb = ve_b[0];
  const int lane = tid & 31;
  const int col = lane & 15;
  const int rowbase = (lane >> 4) << 3;
  const int mslice = (w & 15) * 16 + col;   // fixed m column (waves 0-15)
  float creg[8];                            // cell state, register-resident
#pragma unroll
  for (int e = 0; e < 8; ++e) creg[e] = 0.f;

  for (int t = 0; t < T_; ++t) {
    // ---- Phase 1: we = [h|c] @ We_w^T + We_b  (waves 0-15, one T col-tile each)
    if (w < 16) {
      v8f c = {};
      for (int kk = 0; kk < 16; ++kk) {                 // K = 512
        const _Float16* abase = (kk < 8) ? &sh_h[0][0] : &sh_c[0][0];
        const int kb = (kk & 7) * 32;
        v16h a = load_frag_a([&](int r, int k) -> _Float16 {
          return abase[r * M_ + kb + k];
        });
        v16h bb = load_frag_b_packed(pWe + (size_t)(w * 16 + kk) * 512);
        c = WMMA_F16(a, bb, c);
      }
      store_frag_c(c, [&](int r, int cc, float v) {
        const int tt = w * 16 + cc;
        sh_we[r][tt] = v + We_b[tt];
      });
    }
    __syncthreads();

    // ---- Phase 2: e[r,n] = sum_t' tanh(we[r,t'] + ue_x[b,n,t']) * ve[t']
    for (int p = 0; p < 2; ++p) {
      const int idx = tid + p * 1024;   // 0..2047
      const int r = idx >> 7, n = idx & 127;
      const _Float16* ue = ue_x + ((size_t)(b0 + r) * N_ + n) * T_;
      __builtin_prefetch(ue, 0, 0);     // global_prefetch_b8; ue_x is L2-resident
      float acc = 0.f;
#pragma unroll 8
      for (int tt = 0; tt < T_; ++tt)
        acc += fast_tanh(sh_we[r][tt] + (float)ue[tt]) * sh_ve[tt];
      sh_sc[r][n] = acc + veb;
    }
    __syncthreads();

    // ---- Phase 3: softmax over n (one thread per batch row), then x_tilde
    if (tid < 16) {
      float mx = -1e30f;
      for (int n = 0; n < N_; ++n) mx = fmaxf(mx, sh_sc[tid][n]);
      float s = 0.f;
      for (int n = 0; n < N_; ++n) {
        float e = __expf(sh_sc[tid][n] - mx);
        sh_sc[tid][n] = e;
        s += e;
      }
      const float inv = 1.0f / s;
      for (int n = 0; n < N_; ++n) sh_sc[tid][n] *= inv;
    }
    __syncthreads();
    for (int p = 0; p < 2; ++p) {
      const int idx = tid + p * 1024;
      const int r = idx >> 7, n = idx & 127;
      sh_xt[r][n] =
          (_Float16)(sh_sc[r][n] * x[((size_t)(b0 + r) * T_ + t) * N_ + n]);
    }
    __syncthreads();

    // ---- Phase 4a: gates (i,f,g,o) for this wave's 16-col m slice, in regs
    v8f acc[4];
    if (w < 16) {
      for (int g4 = 0; g4 < 4; ++g4) {
        const int ctile = g4 * 16 + w;                  // gate col-tile 0..63
        v8f c = {};
        for (int kk = 0; kk < N_ / 32; ++kk) {          // x_tilde @ w_ih^T
          const int kb = kk * 32;
          v16h a = load_frag_a([&](int r, int k) -> _Float16 {
            return sh_xt[r][kb + k];
          });
          v16h bb = load_frag_b_packed(pWih + (size_t)(ctile * 4 + kk) * 512);
          c = WMMA_F16(a, bb, c);
        }
        for (int kk = 0; kk < M_ / 32; ++kk) {          // + h @ w_hh^T
          const int kb = kk * 32;
          v16h a = load_frag_a([&](int r, int k) -> _Float16 {
            return sh_h[r][kb + k];
          });
          v16h bb = load_frag_b_packed(pWhh + (size_t)(ctile * 8 + kk) * 512);
          c = WMMA_F16(a, bb, c);
        }
        acc[g4] = c;
      }
    }
    __syncthreads();  // all waves done reading sh_h before it is overwritten

    // ---- Phase 4b: fused LSTM pointwise in registers, update h/c
    if (w < 16) {
      const float bi = sh_bias[mslice];
      const float bf = sh_bias[M_ + mslice];
      const float bg = sh_bias[2 * M_ + mslice];
      const float bo = sh_bias[3 * M_ + mslice];
#pragma unroll
      for (int e = 0; e < 8; ++e) {
        const int r = rowbase + e;
        const float gi = fast_sig(acc[0][e] + bi);
        const float gf = fast_sig(acc[1][e] + bf);
        const float gg = fast_tanh(acc[2][e] + bg);
        const float go = fast_sig(acc[3][e] + bo);
        creg[e] = gf * creg[e] + gi * gg;
        const float hn = go * fast_tanh(creg[e]);
        sh_h[r][mslice] = (_Float16)hn;
        sh_c[r][mslice] = (_Float16)creg[e];
        h_cat[((size_t)(b0 + r) * T_ + t) * M_ + mslice] = (_Float16)hn;
      }
    }
    __syncthreads();
  }
}

// ============================================================================
// Kernel 3a: l[b,t] = sum_m tanh((H_b @ Ud^T)[t,m] + Ud_b[m]) * vd[m] + vd_b
// ============================================================================
__global__ __launch_bounds__(256) void lscore_kernel(
    const _Float16* __restrict__ h_cat, const _Float16* __restrict__ pUd,
    const float* __restrict__ Ud_b, const float* __restrict__ vd_w,
    const float* __restrict__ vd_b, float* __restrict__ lws) {
  __shared__ float sh_l[16];
  const int tid = threadIdx.x;
  const int b = blockIdx.x >> 4;
  const int ttile = blockIdx.x & 15;
  if (tid < 16) sh_l[tid] = 0.f;
  __syncthreads();

  const int w = tid >> 5;
  const _Float16* hb = h_cat + ((size_t)b * T_ + ttile * 16) * M_;
  float pl[8];
#pragma unroll
  for (int i = 0; i < 8; ++i) pl[i] = 0.f;

  for (int j = 0; j < 2; ++j) {
    const int mt = w + j * 8;
    v8f c = {};
    for (int kk = 0; kk < M_ / 32; ++kk) {
      const int kb = kk * 32;
      v16h a = load_frag_a([&](int r, int k) -> _Float16 {
        return hb[(size_t)r * M_ + kb + k];
      });
      v16h bb = load_frag_b_packed(pUd + (size_t)(mt * 8 + kk) * 512);
      c = WMMA_F16(a, bb, c);
    }
    const int m = mt * 16 + (tid & 15);
    const float ub = Ud_b[m], vw = vd_w[m];
#pragma unroll
    for (int i = 0; i < 8; ++i) pl[i] += fast_tanh(c[i] + ub) * vw;
  }
  const int rowbase = ((tid & 31) >> 4) << 3;
#pragma unroll
  for (int i = 0; i < 8; ++i) atomicAdd(&sh_l[rowbase + i], pl[i]);
  __syncthreads();
  if (tid < 16) lws[(size_t)b * T_ + ttile * 16 + tid] = sh_l[tid] + vd_b[0];
}

// ============================================================================
// Kernel 3b: softmax over T, context = beta . H, logits = context . out_w
// ============================================================================
__global__ __launch_bounds__(256) void out_kernel(
    const _Float16* __restrict__ h_cat, const float* __restrict__ lws,
    const float* __restrict__ out_w, const float* __restrict__ out_b,
    float* __restrict__ out_logits, float* __restrict__ out_beta) {
  __shared__ float sh_red[256];
  __shared__ float sh_beta[256];
  const int b = blockIdx.x;
  const int t = threadIdx.x;

  const float lv = lws[(size_t)b * T_ + t];
  sh_red[t] = lv;
  __syncthreads();
  for (int s = 128; s > 0; s >>= 1) {
    if (t < s) sh_red[t] = fmaxf(sh_red[t], sh_red[t + s]);
    __syncthreads();
  }
  const float mx = sh_red[0];
  __syncthreads();
  const float e = __expf(lv - mx);
  sh_red[t] = e;
  __syncthreads();
  for (int s = 128; s > 0; s >>= 1) {
    if (t < s) sh_red[t] += sh_red[t + s];
    __syncthreads();
  }
  const float beta = e / sh_red[0];
  sh_beta[t] = beta;
  out_beta[(size_t)b * T_ + t] = beta;
  __syncthreads();

  const _Float16* hb = h_cat + (size_t)b * T_ * M_;
  float acc = 0.f;
#pragma unroll 4
  for (int tt = 0; tt < T_; ++tt) acc += sh_beta[tt] * (float)hb[(size_t)tt * M_ + t];

  sh_red[t] = acc * out_w[t];
  __syncthreads();
  for (int s = 128; s > 0; s >>= 1) {
    if (t < s) sh_red[t] += sh_red[t + s];
    __syncthreads();
  }
  if (t == 0) out_logits[b] = sh_red[0] + out_b[0];
}

// ============================================================================
extern "C" void kernel_launch(void* const* d_in, const int* in_sizes, int n_in,
                              void* d_out, int out_size, void* d_ws,
                              size_t ws_size, hipStream_t stream) {
  const float* x    = (const float*)d_in[0];
  const float* We_w = (const float*)d_in[1];
  const float* We_b = (const float*)d_in[2];
  const float* Ue_w = (const float*)d_in[3];
  const float* Ue_b = (const float*)d_in[4];
  const float* ve_w = (const float*)d_in[5];
  const float* ve_b = (const float*)d_in[6];
  const float* Ud_w = (const float*)d_in[7];
  const float* Ud_b = (const float*)d_in[8];
  const float* vd_w = (const float*)d_in[9];
  const float* vd_b = (const float*)d_in[10];
  const float* out_w = (const float*)d_in[11];
  const float* out_b = (const float*)d_in[12];
  const float* w_ih = (const float*)d_in[13];
  const float* w_hh = (const float*)d_in[14];
  const float* b_ih = (const float*)d_in[15];
  const float* b_hh = (const float*)d_in[16];

  char* ws = (char*)d_ws;
  size_t o = 0;
  _Float16* ue_x  = (_Float16*)(ws + o); o += 67108864;    //  64 MB, L2-resident
  _Float16* h_cat = (_Float16*)(ws + o); o += 134217728;   // 128 MB
  float*    lws   = (float*)(ws + o);    o += 1048576;     //   1 MB
  _Float16* pUe   = (_Float16*)(ws + o); o += 131072;      // packed f16 weights
  _Float16* pWe   = (_Float16*)(ws + o); o += 262144;
  _Float16* pWih  = (_Float16*)(ws + o); o += 262144;
  _Float16* pWhh  = (_Float16*)(ws + o); o += 524288;
  _Float16* pUd   = (_Float16*)(ws + o); o += 131072;

  float* out_logits = (float*)d_out;       // [B]
  float* out_beta   = (float*)d_out + B_;  // [B, T]

  // 0) pack weights into f16 WMMA-B-fragment order (once per launch, tiny)
  pack_b_kernel<<<dim3(16), dim3(256), 0, stream>>>(Ue_w, pUe, 16, 8, T_);
  pack_b_kernel<<<dim3(32), dim3(256), 0, stream>>>(We_w, pWe, 16, 16, 2 * M_);
  pack_b_kernel<<<dim3(32), dim3(256), 0, stream>>>(w_ih, pWih, 64, 4, N_);
  pack_b_kernel<<<dim3(64), dim3(256), 0, stream>>>(w_hh, pWhh, 64, 8, M_);
  pack_b_kernel<<<dim3(16), dim3(256), 0, stream>>>(Ud_w, pUd, 16, 8, M_);

  // 1) ue_x precompute (TDM-staged A tiles, packed-f16 B)
  ue_gemm_kernel<<<dim3(16384), dim3(256), 0, stream>>>(x, pUe, Ue_b, ue_x);

  // 2) sequential DA-RNN recurrence: 64 blocks x 16 batch rows
  darnn_recurrence_kernel<<<dim3(B_ / 16), dim3(1024), 0, stream>>>(
      x, pWe, We_b, ve_w, ve_b, pWih, pWhh, b_ih, b_hh, ue_x, h_cat);

  // 3) temporal attention scores
  lscore_kernel<<<dim3(B_ * (T_ / 16)), dim3(256), 0, stream>>>(
      h_cat, pUd, Ud_b, vd_w, vd_b, lws);

  // 4) softmax + context + logits
  out_kernel<<<dim3(B_), dim3(256), 0, stream>>>(h_cat, lws, out_w, out_b,
                                                 out_logits, out_beta);
}
